// SABlock_63668595196453
// MI455X (gfx1250) — matrix operand order repaired and checked by reading, not compiled, not run on hardware
//
#include <hip/hip_runtime.h>
#include <hip/hip_bf16.h>

// ---------------------------------------------------------------------------
// CDNA5 (gfx1250) fused multi-head attention with decomposed rel-pos bias.
// GEMMs: v_wmma_f32_16x16x32_bf16, software-pipelined (2 fragment sets live).
// K/V staging: global_load_async_to_lds_b128 (ASYNCcnt) double-buffered.
// ---------------------------------------------------------------------------

typedef __attribute__((ext_vector_type(16))) __bf16 v16bf;
typedef __attribute__((ext_vector_type(8)))  float  v8f;

#define NUM_HEADS 12
#define HEAD_DIM  64
#define IMG_H     32
#define IMG_W     32
#define BATCH     4
#define SEQ       1024          // IMG_H * IMG_W
#define CDIM      768           // NUM_HEADS * HEAD_DIM
#define QKVDIM    2304          // 3 * CDIM

static __device__ __forceinline__ v8f vzero8() {
  v8f z = {0.f, 0.f, 0.f, 0.f, 0.f, 0.f, 0.f, 0.f};
  return z;
}

static __device__ __forceinline__ v8f wmma_bf16(v16bf a, v16bf b, v8f c) {
  // 8 args: (neg_a, A, neg_b, B, c_mod, C, reuse_a, reuse_b)
  return __builtin_amdgcn_wmma_f32_16x16x32_bf16(false, a, false, b, (short)0, c,
                                                 false, false);
}

// Wave-wide async copy global->LDS, 16B per lane, tracked by ASYNCcnt.
static __device__ __forceinline__ void async_ld_b128(unsigned lds_off, const void* g) {
  asm volatile("global_load_async_to_lds_b128 %0, %1, off"
               :: "v"(lds_off), "v"((unsigned long long)(uintptr_t)g)
               : "memory");
}
static __device__ __forceinline__ void wait_async0() {
  asm volatile("s_wait_asynccnt 0x0" ::: "memory");
}

// A-frag (16x32): lane half h, m=lane&15, elem j -> K = h*16 + j
static __device__ __forceinline__ void load_a_frag(const __bf16* __restrict__ X,
                                                   int mA, int kAh, int k0, v16bf& a) {
#pragma unroll
  for (int j = 0; j < 16; ++j)
    a[j] = X[(size_t)mA * CDIM + k0 + kAh + j];
}

// Four B-frags (32x16 each): n=lane&15, K = (lane>>4)*16 + j ; B[k][n] = W[n][k]
static __device__ __forceinline__ void load_b_frag4(const __bf16* __restrict__ W,
                                                    int col0, int nB, int kBh, int k0,
                                                    v16bf* b) {
#pragma unroll
  for (int t = 0; t < 4; ++t) {
    const __bf16* wp = W + (size_t)(col0 + t * 16 + nB) * CDIM + k0 + kBh;
#pragma unroll
    for (int j = 0; j < 16; ++j) b[t][j] = wp[j];
  }
}

// ---------------------------------------------------------------------------
// fp32 -> bf16 elementwise conversion
// ---------------------------------------------------------------------------
__global__ void cvt_f32_bf16(const float* __restrict__ src,
                             __bf16* __restrict__ dst, int n) {
  int i = blockIdx.x * blockDim.x + threadIdx.x;
  int stride = gridDim.x * blockDim.x;
  for (; i < n; i += stride) dst[i] = (__bf16)src[i];
}

// ---------------------------------------------------------------------------
// QKV projection: (4096 x 768) @ (768 x 2304) -> scatter into q/k/v
// layout (B, nh, N, hd), bf16.  One 16x64 output tile per wave,
// software-pipelined over K (two fragment sets in flight).
// ---------------------------------------------------------------------------
__global__ void qkv_gemm(const __bf16* __restrict__ X, const __bf16* __restrict__ W,
                         __bf16* __restrict__ Q, __bf16* __restrict__ K,
                         __bf16* __restrict__ V) {
  const int lane = threadIdx.x & 31;
  const int wv = threadIdx.x >> 5;
  const int wg = blockIdx.x * 8 + wv;        // 9216 wave-tiles total
  const int cg = wg % (QKVDIM / 64);
  const int rt = wg / (QKVDIM / 64);
  const int row0 = rt * 16;
  const int col0 = cg * 64;

  v8f acc[4];
#pragma unroll
  for (int t = 0; t < 4; ++t) acc[t] = vzero8();

  const int mA = row0 + (lane & 15);
  const int kAh = (lane >> 4) << 4;
  const int nB = lane & 15;
  const int kBh = (lane >> 4) << 4;

  v16bf a0, a1, b0[4], b1[4];
  load_a_frag(X, mA, kAh, 0, a0);
  load_b_frag4(W, col0, nB, kBh, 0, b0);

#pragma unroll 2
  for (int k0 = 0; k0 < CDIM; k0 += 64) {
    // issue loads for k0+32 while consuming k0
    load_a_frag(X, mA, kAh, k0 + 32, a1);
    load_b_frag4(W, col0, nB, kBh, k0 + 32, b1);
#pragma unroll
    for (int t = 0; t < 4; ++t) acc[t] = wmma_bf16(a0, b0[t], acc[t]);
    // issue loads for k0+64 while consuming k0+32
    if (k0 + 64 < CDIM) {
      load_a_frag(X, mA, kAh, k0 + 64, a0);
      load_b_frag4(W, col0, nB, kBh, k0 + 64, b0);
    }
#pragma unroll
    for (int t = 0; t < 4; ++t) acc[t] = wmma_bf16(a1, b1[t], acc[t]);
  }

#pragma unroll
  for (int t = 0; t < 4; ++t) {
#pragma unroll
    for (int r = 0; r < 8; ++r) {
      int m = row0 + ((lane >> 4) << 3) + r;
      int c = col0 + t * 16 + (lane & 15);
      int which = c / CDIM;                  // 0=q, 1=k, 2=v
      int rem = c - which * CDIM;
      int h = rem >> 6, d = rem & 63;
      int b_ = m >> 10, nq = m & 1023;
      __bf16* dst = (which == 0) ? Q : (which == 1) ? K : V;
      dst[((((size_t)b_ * NUM_HEADS + h) << 10) + nq) * HEAD_DIM + d] =
          (__bf16)acc[t][r];
    }
  }
}

// ---------------------------------------------------------------------------
// Fused attention, flash-style online softmax with double-buffered
// ASYNC global->LDS staging of K/V blocks.
// Block = 256 threads (8 waves) = one (b,h) x 128 q-rows; wave owns 16 rows.
// ---------------------------------------------------------------------------
__global__ void attention(const __bf16* __restrict__ Q, const __bf16* __restrict__ K,
                          const __bf16* __restrict__ V, const float* __restrict__ RPH,
                          const float* __restrict__ RPW, __bf16* __restrict__ O) {
  __shared__ __align__(16) __bf16 Ks[2][128 * 64];    // 32 KB (double buffer)
  __shared__ __align__(16) __bf16 Vs[2][128 * 64];    // 32 KB
  __shared__ __align__(16) __bf16 Ps[8][16 * 128];    // 32 KB (per-wave P scratch)
  __shared__ float Bhs[8][16][32];                    // 16 KB
  __shared__ float Bws[8][16][32];                    // 16 KB -> 128 KB (<320KB/WGP)

  const int lane = threadIdx.x & 31;
  const int wv = threadIdx.x >> 5;
  const int tid = threadIdx.x;
  const int bhead = blockIdx.x >> 3;   // 0..47  (b*12+h)
  const int qb = blockIdx.x & 7;       // q-block of 128 rows
  const int bglob = bhead / NUM_HEADS;
  const int h = bhead % NUM_HEADS;
  const int qrow0 = qb * 128 + wv * 16;

  const __bf16* qg = Q + (size_t)bhead * SEQ * HEAD_DIM;
  const __bf16* kg = K + (size_t)bhead * SEQ * HEAD_DIM;
  const __bf16* vg = V + (size_t)bhead * SEQ * HEAD_DIM;

  // ---- prologue: issue async staging of key-block 0 into buffer 0 ---------
#pragma unroll
  for (int i = tid; i < 1024; i += 256) {
    async_ld_b128((unsigned)(uintptr_t)(&Ks[0][i * 8]), kg + (size_t)i * 8);
    async_ld_b128((unsigned)(uintptr_t)(&Vs[0][i * 8]), vg + (size_t)i * 8);
  }

  // ---- decomposed rel-pos bias tables for this wave's 16 rows -------------
  for (int idx = lane; idx < 512; idx += 32) {
    int i = idx >> 5, t = idx & 31;
    int ng = qrow0 + i;
    int hq = ng >> 5, wq = ng & 31;
    const __bf16* qr = qg + (size_t)ng * HEAD_DIM;
    const float* rh = RPH + (size_t)(hq - t + IMG_H - 1) * HEAD_DIM;
    const float* rw = RPW + (size_t)(wq - t + IMG_W - 1) * HEAD_DIM;
    float sh = 0.f, sw = 0.f;
#pragma unroll 8
    for (int d = 0; d < HEAD_DIM; ++d) {
      float qd = (float)qr[d];
      sh += qd * rh[d];
      sw += qd * rw[d];
    }
    Bhs[wv][i][t] = sh;
    Bws[wv][i][t] = sw;
  }

  // ---- Q A-fragments (d = 64 -> two 16x32 fragments), resident ------------
  v16bf qa[2];
  {
    int m = qrow0 + (lane & 15);
    int kh = (lane >> 4) << 4;
#pragma unroll
    for (int s = 0; s < 2; ++s)
#pragma unroll
      for (int j = 0; j < 16; ++j)
        qa[s][j] = qg[(size_t)m * HEAD_DIM + s * 32 + kh + j];
  }

  v8f o[4];
#pragma unroll
  for (int t = 0; t < 4; ++t) o[t] = vzero8();
  float mm[8], ll[8];
#pragma unroll
  for (int r = 0; r < 8; ++r) { mm[r] = -1e30f; ll[r] = 0.f; }

  const float scale = 0.125f;  // HEAD_DIM^-0.5

  for (int kb = 0; kb < 8; ++kb) {
    const int cur = kb & 1;
    // Current buffer becomes valid; everyone done reading the buffer we are
    // about to overwrite with the next prefetch.
    wait_async0();
    __syncthreads();
    if (kb < 7) {
      const int nxt = 1 - cur;
      const __bf16* kn = kg + (size_t)(kb + 1) * 128 * 64;
      const __bf16* vn = vg + (size_t)(kb + 1) * 128 * 64;
#pragma unroll
      for (int i = tid; i < 1024; i += 256) {
        async_ld_b128((unsigned)(uintptr_t)(&Ks[nxt][i * 8]), kn + (size_t)i * 8);
        async_ld_b128((unsigned)(uintptr_t)(&Vs[nxt][i * 8]), vn + (size_t)i * 8);
      }
    }

    // ---- S = q @ K^T : 16 WMMAs (8 col-tiles x 2 d-steps) -----------------
    v8f s[8];
#pragma unroll
    for (int nt = 0; nt < 8; ++nt) {
      s[nt] = vzero8();
      int key = nt * 16 + (lane & 15);
#pragma unroll
      for (int ks_ = 0; ks_ < 2; ++ks_) {
        int kh = ks_ * 32 + ((lane >> 4) << 4);
        v16bf b;
#pragma unroll
        for (int j = 0; j < 16; ++j) b[j] = Ks[cur][key * 64 + kh + j];
        s[nt] = wmma_bf16(qa[ks_], b, s[nt]);
      }
    }

    // ---- scale + rel-pos bias --------------------------------------------
#pragma unroll
    for (int nt = 0; nt < 8; ++nt) {
      int nk = kb * 128 + nt * 16 + (lane & 15);
      int hk = nk >> 5, wk = nk & 31;
#pragma unroll
      for (int r = 0; r < 8; ++r) {
        int mi = ((lane >> 4) << 3) + r;
        s[nt][r] = s[nt][r] * scale + Bhs[wv][mi][hk] + Bws[wv][mi][wk];
      }
    }

    // ---- online softmax (per-row, rows live in 16-lane halves) -----------
#pragma unroll
    for (int r = 0; r < 8; ++r) {
      float cand = s[0][r];
#pragma unroll
      for (int nt = 1; nt < 8; ++nt) cand = fmaxf(cand, s[nt][r]);
#pragma unroll
      for (int off = 1; off < 16; off <<= 1)
        cand = fmaxf(cand, __shfl_xor(cand, off, 32));
      float newm = fmaxf(mm[r], cand);
      float corr = __expf(mm[r] - newm);
      mm[r] = newm;
      float rsum = 0.f;
#pragma unroll
      for (int nt = 0; nt < 8; ++nt) {
        float p = __expf(s[nt][r] - newm);
        s[nt][r] = p;
        rsum += p;
      }
#pragma unroll
      for (int off = 1; off < 16; off <<= 1) rsum += __shfl_xor(rsum, off, 32);
      ll[r] = ll[r] * corr + rsum;
#pragma unroll
      for (int ot = 0; ot < 4; ++ot) o[ot][r] *= corr;
    }

    // ---- P (C-layout) -> LDS row-major bf16, reload as A-fragments --------
#pragma unroll
    for (int nt = 0; nt < 8; ++nt)
#pragma unroll
      for (int r = 0; r < 8; ++r)
        Ps[wv][(((lane >> 4) << 3) + r) * 128 + nt * 16 + (lane & 15)] =
            (__bf16)s[nt][r];

    // ---- O += P @ V : 16 WMMAs (4 k-steps x 4 d-tiles) --------------------
#pragma unroll
    for (int kst = 0; kst < 4; ++kst) {
      v16bf pa;
      {
        int m = lane & 15;
        int kh = (lane >> 4) << 4;
#pragma unroll
        for (int j = 0; j < 16; ++j)
          pa[j] = Ps[wv][m * 128 + kst * 32 + kh + j];
      }
#pragma unroll
      for (int dt = 0; dt < 4; ++dt) {
        int d = dt * 16 + (lane & 15);
        int kh = kst * 32 + ((lane >> 4) << 4);
        v16bf b;
#pragma unroll
        for (int j = 0; j < 16; ++j) b[j] = Vs[cur][(kh + j) * 64 + d];
        o[dt] = wmma_bf16(pa, b, o[dt]);
      }
    }
  }

  // ---- normalize and store to (B, N, C) bf16 attn buffer -------------------
#pragma unroll
  for (int dt = 0; dt < 4; ++dt)
#pragma unroll
    for (int r = 0; r < 8; ++r) {
      int mi = ((lane >> 4) << 3) + r;
      int ng = qrow0 + mi;
      int d = dt * 16 + (lane & 15);
      float val = o[dt][r] / ll[r];
      O[((size_t)bglob * SEQ + ng) * CDIM + h * HEAD_DIM + d] = (__bf16)val;
    }
}

// ---------------------------------------------------------------------------
// Output projection: (4096 x 768) @ (768 x 768) + bias -> fp32 out,
// software-pipelined like qkv_gemm.
// ---------------------------------------------------------------------------
__global__ void out_gemm(const __bf16* __restrict__ A, const __bf16* __restrict__ W,
                         const float* __restrict__ bias, float* __restrict__ Out) {
  const int lane = threadIdx.x & 31;
  const int wv = threadIdx.x >> 5;
  const int wg = blockIdx.x * 8 + wv;        // 3072 wave-tiles
  const int cg = wg % (CDIM / 64);
  const int rt = wg / (CDIM / 64);
  const int row0 = rt * 16;
  const int col0 = cg * 64;

  v8f acc[4];
#pragma unroll
  for (int t = 0; t < 4; ++t) acc[t] = vzero8();

  const int mA = row0 + (lane & 15);
  const int kAh = (lane >> 4) << 4;
  const int nB = lane & 15;
  const int kBh = (lane >> 4) << 4;

  v16bf a0, a1, b0[4], b1[4];
  load_a_frag(A, mA, kAh, 0, a0);
  load_b_frag4(W, col0, nB, kBh, 0, b0);

#pragma unroll 2
  for (int k0 = 0; k0 < CDIM; k0 += 64) {
    load_a_frag(A, mA, kAh, k0 + 32, a1);
    load_b_frag4(W, col0, nB, kBh, k0 + 32, b1);
#pragma unroll
    for (int t = 0; t < 4; ++t) acc[t] = wmma_bf16(a0, b0[t], acc[t]);
    if (k0 + 64 < CDIM) {
      load_a_frag(A, mA, kAh, k0 + 64, a0);
      load_b_frag4(W, col0, nB, kBh, k0 + 64, b0);
    }
#pragma unroll
    for (int t = 0; t < 4; ++t) acc[t] = wmma_bf16(a1, b1[t], acc[t]);
  }

#pragma unroll
  for (int t = 0; t < 4; ++t) {
#pragma unroll
    for (int r = 0; r < 8; ++r) {
      int m = row0 + ((lane >> 4) << 3) + r;
      int c = col0 + t * 16 + (lane & 15);
      Out[(size_t)m * CDIM + c] = acc[t][r] + bias[c];
    }
  }
}

// ---------------------------------------------------------------------------
// Host launcher. Workspace layout (bytes, all regions fully written each call):
//   xbf     @ 0          6,291,456   (4096x768 bf16)
//   wqkv_bf @ 6291456    3,538,944   (2304x768 bf16)
//   wout_bf @ 9830400    1,179,648   (768x768 bf16)
//   q       @ 11010048   6,291,456   (4,12,1024,64 bf16)
//   k       @ 17301504   6,291,456
//   v       @ 23592960   6,291,456
//   attn    @ 29884416   6,291,456   (4096x768 bf16)
// ---------------------------------------------------------------------------
extern "C" void kernel_launch(void* const* d_in, const int* in_sizes, int n_in,
                              void* d_out, int out_size, void* d_ws, size_t ws_size,
                              hipStream_t stream) {
  const float* x    = (const float*)d_in[0];
  const float* Wqkv = (const float*)d_in[1];
  const float* Wout = (const float*)d_in[2];
  const float* bout = (const float*)d_in[3];
  const float* rph  = (const float*)d_in[4];
  const float* rpw  = (const float*)d_in[5];
  float* out = (float*)d_out;

  char* ws = (char*)d_ws;
  __bf16* xbf     = (__bf16*)(ws + 0);
  __bf16* wqkv_bf = (__bf16*)(ws + 6291456);
  __bf16* wout_bf = (__bf16*)(ws + 9830400);
  __bf16* qb      = (__bf16*)(ws + 11010048);
  __bf16* kb      = (__bf16*)(ws + 17301504);
  __bf16* vb      = (__bf16*)(ws + 23592960);
  __bf16* attn    = (__bf16*)(ws + 29884416);

  cvt_f32_bf16<<<1024, 256, 0, stream>>>(x, xbf, BATCH * SEQ * CDIM);
  cvt_f32_bf16<<<1024, 256, 0, stream>>>(Wqkv, wqkv_bf, QKVDIM * CDIM);
  cvt_f32_bf16<<<512, 256, 0, stream>>>(Wout, wout_bf, CDIM * CDIM);

  // 256 row-tiles x 36 col-groups = 9216 wave-tiles / 8 waves per block
  qkv_gemm<<<1152, 256, 0, stream>>>(xbf, wqkv_bf, qb, kb, vb);

  // 48 (b,h) pairs x 8 q-blocks of 128 rows
  attention<<<384, 256, 0, stream>>>(qb, kb, vb, rph, rpw, attn);

  // 256 row-tiles x 12 col-groups = 3072 wave-tiles / 8 per block
  out_gemm<<<384, 256, 0, stream>>>(attn, wout_bf, bout, out);
}
